// MultiHeadAttention_85959475462605
// MI455X (gfx1250) — matrix operand order repaired and checked by reading, not compiled, not run on hardware
//
#include <hip/hip_runtime.h>
#include <hip/hip_bf16.h>

typedef __attribute__((ext_vector_type(16))) _Float16 v16h;
typedef __attribute__((ext_vector_type(8)))  _Float16 v8h;
typedef __attribute__((ext_vector_type(8)))  float    v8f;
typedef __attribute__((ext_vector_type(4)))  int      v4i;

#define D_MODEL   1024
#define NUM_HEADS 16
#define DEPTH     64
#define BATCH     2
#define SEQ       2048

// ---------------------------------------------------------------------------
// Async global->LDS 16-byte copy (GLOBAL_LOAD_ASYNC_TO_LDS_B128, ASYNCcnt).
// Builtin exists on this toolchain with v4i-typed pointer params.
// ---------------------------------------------------------------------------
__device__ __forceinline__ void async_copy_b128(const _Float16* g, _Float16* l)
{
#if __has_builtin(__builtin_amdgcn_global_load_async_to_lds_b128)
    __attribute__((address_space(1))) void* g1 =
        (__attribute__((address_space(1))) void*)g;
    __attribute__((address_space(3))) void* l3 =
        (__attribute__((address_space(3))) void*)l;
    __builtin_amdgcn_global_load_async_to_lds_b128(
        (__attribute__((address_space(1))) v4i*)g1,
        (__attribute__((address_space(3))) v4i*)l3, 0, 0);
#else
    unsigned ldsoff = (unsigned)(unsigned long long)
        (__attribute__((address_space(3))) _Float16*)l;
    asm volatile("global_load_async_to_lds_b128 %0, %1, off"
                 :: "v"(ldsoff), "v"(g) : "memory");
#endif
}

__device__ __forceinline__ void async_wait_all()
{
#if __has_builtin(__builtin_amdgcn_s_wait_asynccnt)
    __builtin_amdgcn_s_wait_asynccnt(0);
#else
    asm volatile("s_wait_asynccnt 0x0" ::: "memory");
#endif
}

// ---------------------------------------------------------------------------
// f32 -> f16 conversion, 8 elements per thread, vector b128 store.
// ---------------------------------------------------------------------------
__global__ __launch_bounds__(256)
void cvt_f16_kernel(const float* __restrict__ src, _Float16* __restrict__ dst,
                    long n8)
{
    long i = (long)blockIdx.x * blockDim.x + threadIdx.x;
    if (i >= n8) return;
    const float* s = src + i * 8;
    v8h o;
    #pragma unroll
    for (int e = 0; e < 8; ++e) o[e] = (_Float16)s[e];
    *reinterpret_cast<v8h*>(dst + i * 8) = o;
}

// ---------------------------------------------------------------------------
// Projection GEMM: Y = X @ W^T + b   (X:[4096,1024] f16, W:[1024,1024] f16)
// All 8 waves of a block share one 16-row A strip: staged once in LDS via
// async global->LDS DMA; A-frags then come from ds_load_b128, only B streams
// from global.  Output f16 head-split:
//   transposed==0 : Y[b][h][s][d]   ([B,H,S,D])
//   transposed==1 : Y[b][h][d][s]   ([B,H,D,S])  (for V: AV B-frags contiguous)
// ---------------------------------------------------------------------------
__global__ __launch_bounds__(256)
void proj_gemm_kernel(const _Float16* __restrict__ X,
                      const _Float16* __restrict__ W,
                      const float* __restrict__ bias,
                      _Float16* __restrict__ Y,
                      int transposed)
{
    __shared__ __align__(32) _Float16 s_a[16 * D_MODEL];   // 32 KB A strip

    const int tid  = threadIdx.x;
    const int lane = tid & 31;
    const int wave = tid >> 5;
    const int tileRow = blockIdx.x >> 3;            // same for all 8 waves
    const int tileCol = ((blockIdx.x & 7) << 3) + wave;
    const int M  = lane & 15;
    const int hi = lane >> 4;

    // Stage A strip (16 x 1024 f16) cooperatively: 2048 x 16B async copies.
    const int aRowBase = tileRow * 16;
    #pragma unroll
    for (int i = 0; i < 8; ++i) {
        const int chunk = tid + i * 256;
        const int row = chunk >> 7;                 // 128 chunks per row
        const int col = (chunk & 127) * 8;
        async_copy_b128(X + (long)(aRowBase + row) * D_MODEL + col,
                        s_a + row * D_MODEL + col);
    }
    async_wait_all();
    __syncthreads();

    const int bCol = tileCol * 16 + M;              // B-frag col (N = lane&15)
    const _Float16* aL    = s_a + M * D_MODEL + hi * 8;
    const _Float16* bBase = W + (long)bCol * D_MODEL + hi * 8;

    v8f c = {};
    for (int k0 = 0; k0 < D_MODEL; k0 += 32) {
        // A (16x32 f16): e -> K = (e&7) + (e>=8)*16 + hi*8  => two LDS b128
        v8h alo = *reinterpret_cast<const v8h*>(aL + k0);
        v8h ahi = *reinterpret_cast<const v8h*>(aL + k0 + 16);
        v16h a;
        #pragma unroll
        for (int e = 0; e < 8; ++e) { a[e] = alo[e]; a[e + 8] = ahi[e]; }
        // B (32x16 f16): e -> K = (e&7) + (e>=8)*16 + hi*8  => two global b128
        v8h blo = *reinterpret_cast<const v8h*>(bBase + k0);
        v8h bhi = *reinterpret_cast<const v8h*>(bBase + k0 + 16);
        v16h b;
        #pragma unroll
        for (int e = 0; e < 8; ++e) { b[e] = blo[e]; b[e + 8] = bhi[e]; }
        __builtin_prefetch((const void*)(bBase + k0 + 32), 0, 0);
        c = __builtin_amdgcn_wmma_f32_16x16x32_f16(false, a, false, b,
                                                   (short)0, c, false, false);
    }

    const float bval = bias[bCol];
    const int h = bCol >> 6;
    const int d = bCol & 63;
    #pragma unroll
    for (int v = 0; v < 8; ++v) {
        const int r  = tileRow * 16 + hi * 8 + v;
        const int bb = r >> 11;
        const int s  = r & (SEQ - 1);
        const float val = c[v] + bval;
        long idx;
        if (!transposed)
            idx = (((long)(bb * NUM_HEADS + h) * SEQ) + s) * DEPTH + d;
        else
            idx = (((long)(bb * NUM_HEADS + h) * DEPTH) + d) * SEQ + s;
        Y[idx] = (_Float16)val;
    }
}

// ---------------------------------------------------------------------------
// Attention: one block = one 16-query strip of one (b,h).
// Phase 1: scores (16x2048) via WMMA -> LDS f32 strip (128 KB)
// Phase 2: row softmax; stream probs once to global attn (f32) and to an
//          f16 LDS strip for the AV matrix phase.
// Phase 3: ctx (16x64) = probs @ V via WMMA; 8 waves split K in halves,
//          upper-half partials reduced through the (free) f32 strip.
// ---------------------------------------------------------------------------
__global__ __launch_bounds__(256)
void attn_kernel(const _Float16* __restrict__ Qh,   // [B,H,S,D]
                 const _Float16* __restrict__ Kh,   // [B,H,S,D]
                 const _Float16* __restrict__ Vt,   // [B,H,D,S]
                 _Float16* __restrict__ ctx,        // [B,S,D_MODEL] f16
                 float* __restrict__ attn)          // [B,H,S,S] f32
{
    __shared__ float                  s_scores[16 * SEQ];  // 128 KB
    __shared__ __align__(32) _Float16 s_probs[16 * SEQ];   //  64 KB
    __shared__ __align__(32) _Float16 s_q[16 * DEPTH];     //   2 KB
    __shared__ float                  s_red[256];
    __shared__ float                  s_rowmax[16];
    __shared__ float                  s_rowsum[16];

    const int tid   = threadIdx.x;
    const int lane  = tid & 31;
    const int wave  = tid >> 5;
    const int qTile = blockIdx.x & 127;
    const int bh    = blockIdx.x >> 7;
    const int qBase = qTile * 16;

    const _Float16* Q = Qh + (long)bh * SEQ * DEPTH;
    const _Float16* K = Kh + (long)bh * SEQ * DEPTH;
    const _Float16* V = Vt + (long)bh * DEPTH * SEQ;

    // Stage Q strip (16x64 f16) via async DMA (128 x 16B chunks)
    if (tid < 128) {
        const int row = tid >> 3, col = (tid & 7) * 8;
        async_copy_b128(Q + (long)(qBase + row) * DEPTH + col,
                        s_q + row * DEPTH + col);
    }
    async_wait_all();
    __syncthreads();

    const int M  = lane & 15;
    const int hi = lane >> 4;

    // ---- Phase 1: scores = Q K^T / sqrt(Dh) ----
    for (int t = wave; t < SEQ / 16; t += 8) {
        const int kBase = t * 16;
        v8f c = {};
        #pragma unroll
        for (int d0 = 0; d0 < DEPTH; d0 += 32) {
            const _Float16* aq = s_q + M * DEPTH + d0 + hi * 8;
            v8h alo = *reinterpret_cast<const v8h*>(aq);
            v8h ahi = *reinterpret_cast<const v8h*>(aq + 16);
            v16h a;
            #pragma unroll
            for (int e = 0; e < 8; ++e) { a[e] = alo[e]; a[e + 8] = ahi[e]; }
            // B[d][n] = K[kBase+n][d]; same k order as A (two 8-wide runs)
            const _Float16* bk = K + (long)(kBase + M) * DEPTH + d0 + hi * 8;
            v8h blo = *reinterpret_cast<const v8h*>(bk);
            v8h bhi = *reinterpret_cast<const v8h*>(bk + 16);
            v16h b;
            #pragma unroll
            for (int e = 0; e < 8; ++e) { b[e] = blo[e]; b[e + 8] = bhi[e]; }
            c = __builtin_amdgcn_wmma_f32_16x16x32_f16(false, a, false, b,
                                                       (short)0, c, false, false);
        }
        #pragma unroll
        for (int v = 0; v < 8; ++v)
            s_scores[(hi * 8 + v) * SEQ + kBase + M] = c[v] * 0.125f; // 1/sqrt(64)
    }
    __syncthreads();

    // ---- Phase 2: softmax, 16 threads per row ----
    const int r   = tid >> 4;
    const int sub = tid & 15;
    float mx = -1e30f;
    for (int j = sub; j < SEQ; j += 16)
        mx = fmaxf(mx, s_scores[r * SEQ + j]);
    s_red[tid] = mx;
    __syncthreads();
    if (sub == 0) {
        float mm = s_red[tid];
        #pragma unroll
        for (int i = 1; i < 16; ++i) mm = fmaxf(mm, s_red[tid + i]);
        s_rowmax[r] = mm;
    }
    __syncthreads();
    const float rowmax = s_rowmax[r];
    float sum = 0.f;
    for (int j = sub; j < SEQ; j += 16) {
        float e = __expf(s_scores[r * SEQ + j] - rowmax);
        s_scores[r * SEQ + j] = e;
        sum += e;
    }
    s_red[tid] = sum;
    __syncthreads();
    if (sub == 0) {
        float ss = 0.f;
        #pragma unroll
        for (int i = 0; i < 16; ++i) ss += s_red[tid + i];
        s_rowsum[r] = ss;
    }
    __syncthreads();
    const float inv = 1.0f / s_rowsum[r];
    float* attnRow = attn + ((long)bh * SEQ + qBase + r) * SEQ;
    for (int j = sub; j < SEQ; j += 16) {
        float p = s_scores[r * SEQ + j] * inv;
        s_probs[r * SEQ + j] = (_Float16)p;   // f16 copy for AV matrix phase
        attnRow[j] = p;                       // stream 537 MB attn output once
    }
    __syncthreads();

    // ---- Phase 3: ctx = probs @ V ----
    // wave = kh*4 + colT : colT selects 16-col tile of Dh, kh selects K half.
    const int colT = wave & 3;
    const int kh   = wave >> 2;
    const int n0   = colT * 16;
    v8f c = {};
    const _Float16* vB = V + (long)(n0 + M) * SEQ + hi * 8;
    for (int k0 = kh * 1024; k0 < kh * 1024 + 1024; k0 += 32) {
        const _Float16* ap = s_probs + M * SEQ + k0 + hi * 8;
        v8h alo = *reinterpret_cast<const v8h*>(ap);
        v8h ahi = *reinterpret_cast<const v8h*>(ap + 16);
        v16h a;
        #pragma unroll
        for (int e = 0; e < 8; ++e) { a[e] = alo[e]; a[e + 8] = ahi[e]; }
        // B[k][n] = V[s=k][d=n0+n]; Vt is d-major so contiguous along k
        v8h blo = *reinterpret_cast<const v8h*>(vB + k0);
        v8h bhi = *reinterpret_cast<const v8h*>(vB + k0 + 16);
        v16h b;
        #pragma unroll
        for (int e = 0; e < 8; ++e) { b[e] = blo[e]; b[e + 8] = bhi[e]; }
        c = __builtin_amdgcn_wmma_f32_16x16x32_f16(false, a, false, b,
                                                   (short)0, c, false, false);
    }
    // Upper-half waves park partials in the (now free) f32 score strip.
    if (kh == 1) {
        #pragma unroll
        for (int v = 0; v < 8; ++v)
            s_scores[colT * 256 + (hi * 8 + v) * 16 + M] = c[v];
    }
    __syncthreads();
    if (kh == 0) {
        const int b0 = bh >> 4;
        const int h  = bh & 15;
        #pragma unroll
        for (int v = 0; v < 8; ++v) {
            float val = c[v] + s_scores[colT * 256 + (hi * 8 + v) * 16 + M];
            const int s = qBase + hi * 8 + v;
            ctx[((long)b0 * SEQ + s) * D_MODEL + h * DEPTH + n0 + M] =
                (_Float16)val;
        }
    }
}

// ---------------------------------------------------------------------------
// Output projection: out = ctx @ Wo^T + bo  (ctx f16, Wo f16, out f32)
// Same LDS A-strip staging as proj_gemm_kernel.
// ---------------------------------------------------------------------------
__global__ __launch_bounds__(256)
void out_proj_kernel(const _Float16* __restrict__ X,
                     const _Float16* __restrict__ W,
                     const float* __restrict__ bias,
                     float* __restrict__ out)
{
    __shared__ __align__(32) _Float16 s_a[16 * D_MODEL];   // 32 KB A strip

    const int tid  = threadIdx.x;
    const int lane = tid & 31;
    const int wave = tid >> 5;
    const int tileRow = blockIdx.x >> 3;
    const int tileCol = ((blockIdx.x & 7) << 3) + wave;
    const int M  = lane & 15;
    const int hi = lane >> 4;

    const int aRowBase = tileRow * 16;
    #pragma unroll
    for (int i = 0; i < 8; ++i) {
        const int chunk = tid + i * 256;
        const int row = chunk >> 7;
        const int col = (chunk & 127) * 8;
        async_copy_b128(X + (long)(aRowBase + row) * D_MODEL + col,
                        s_a + row * D_MODEL + col);
    }
    async_wait_all();
    __syncthreads();

    const int bCol = tileCol * 16 + M;
    const _Float16* aL    = s_a + M * D_MODEL + hi * 8;
    const _Float16* bBase = W + (long)bCol * D_MODEL + hi * 8;

    v8f c = {};
    for (int k0 = 0; k0 < D_MODEL; k0 += 32) {
        v8h alo = *reinterpret_cast<const v8h*>(aL + k0);
        v8h ahi = *reinterpret_cast<const v8h*>(aL + k0 + 16);
        v16h a;
        #pragma unroll
        for (int e = 0; e < 8; ++e) { a[e] = alo[e]; a[e + 8] = ahi[e]; }
        v8h blo = *reinterpret_cast<const v8h*>(bBase + k0);
        v8h bhi = *reinterpret_cast<const v8h*>(bBase + k0 + 16);
        v16h b;
        #pragma unroll
        for (int e = 0; e < 8; ++e) { b[e] = blo[e]; b[e + 8] = bhi[e]; }
        __builtin_prefetch((const void*)(bBase + k0 + 32), 0, 0);
        c = __builtin_amdgcn_wmma_f32_16x16x32_f16(false, a, false, b,
                                                   (short)0, c, false, false);
    }
    const float bval = bias[bCol];
    #pragma unroll
    for (int v = 0; v < 8; ++v) {
        const int r = tileRow * 16 + hi * 8 + v;
        out[(long)r * D_MODEL + bCol] = c[v] + bval;
    }
}

// ---------------------------------------------------------------------------
extern "C" void kernel_launch(void* const* d_in, const int* in_sizes, int n_in,
                              void* d_out, int out_size, void* d_ws, size_t ws_size,
                              hipStream_t stream) {
    const float* q  = (const float*)d_in[0];
    const float* k  = (const float*)d_in[1];
    const float* v  = (const float*)d_in[2];
    const float* Wq = (const float*)d_in[3];
    const float* bq = (const float*)d_in[4];
    const float* Wk = (const float*)d_in[5];
    const float* bk = (const float*)d_in[6];
    const float* Wv = (const float*)d_in[7];
    const float* bv = (const float*)d_in[8];
    const float* Wo = (const float*)d_in[9];
    const float* bo = (const float*)d_in[10];

    const size_t elems = (size_t)BATCH * SEQ * D_MODEL;  // 4 Mi
    const size_t welems = (size_t)D_MODEL * D_MODEL;     // 1 Mi

    _Float16* p = (_Float16*)d_ws;
    _Float16* qf  = p; p += elems;   // f16 copies of activations
    _Float16* kf  = p; p += elems;
    _Float16* vf  = p; p += elems;
    _Float16* Wqf = p; p += welems;  // f16 copies of weights
    _Float16* Wkf = p; p += welems;
    _Float16* Wvf = p; p += welems;
    _Float16* Wof = p; p += welems;
    _Float16* Qh  = p; p += elems;   // [B,H,S,D]
    _Float16* Kh  = p; p += elems;   // [B,H,S,D]
    _Float16* Vt  = p; p += elems;   // [B,H,D,S]
    _Float16* Ctx = p; p += elems;   // [B,S,1024]

    float* out  = (float*)d_out;     // [B,S,1024] f32
    float* attn = out + elems;       // [B,H,S,S]  f32

    dim3 blk(256);
    // One-time f32 -> f16 conversions (activations + weights)
    cvt_f16_kernel<<<(int)(elems / 8 / 256), blk, 0, stream>>>(q,  qf,  elems / 8);
    cvt_f16_kernel<<<(int)(elems / 8 / 256), blk, 0, stream>>>(k,  kf,  elems / 8);
    cvt_f16_kernel<<<(int)(elems / 8 / 256), blk, 0, stream>>>(v,  vf,  elems / 8);
    cvt_f16_kernel<<<(int)(welems / 8 / 256), blk, 0, stream>>>(Wq, Wqf, welems / 8);
    cvt_f16_kernel<<<(int)(welems / 8 / 256), blk, 0, stream>>>(Wk, Wkf, welems / 8);
    cvt_f16_kernel<<<(int)(welems / 8 / 256), blk, 0, stream>>>(Wv, Wvf, welems / 8);
    cvt_f16_kernel<<<(int)(welems / 8 / 256), blk, 0, stream>>>(Wo, Wof, welems / 8);

    // Projections: 16384 tiles / 8 waves = 2048 blocks each
    proj_gemm_kernel<<<2048, blk, 0, stream>>>(qf, Wqf, bq, Qh, 0);
    proj_gemm_kernel<<<2048, blk, 0, stream>>>(kf, Wkf, bk, Kh, 0);
    proj_gemm_kernel<<<2048, blk, 0, stream>>>(vf, Wvf, bv, Vt, 1);
    // Attention: 32 (b,h) * 128 query strips = 4096 blocks
    attn_kernel<<<4096, blk, 0, stream>>>(Qh, Kh, Vt, Ctx, attn);
    // Output projection
    out_proj_kernel<<<2048, blk, 0, stream>>>(Ctx, Wof, bo, out);
}
